// MultiLayerSRU_37915971289649
// MI455X (gfx1250) — compile-verified, matrix operand order checked by reading, not run on hardware
//
#include <hip/hip_runtime.h>

// ---------------------------------------------------------------------------
// MultiLayerSRU on gfx1250 (MI455X):
//   layer: U = A @ W  (WMMA bf16, f32 accum, TDM double-buffered tiles)
//          -> sequential SRU scan over S
// ---------------------------------------------------------------------------

typedef __attribute__((ext_vector_type(16))) __bf16 v16bf;
typedef __attribute__((ext_vector_type(8)))  __bf16 v8bf;
typedef __attribute__((ext_vector_type(8)))  float  v8f;

#define B_   32
#define S_   1024
#define DIN_ 256
#define H_   512
#define NH_  1536          // 3*H
#define M_   (B_ * S_)     // 32768 rows in the GEMMs

// ---------------- Tensor Data Mover (TDM) support --------------------------

#if __has_builtin(__builtin_amdgcn_tensor_load_to_lds)
#define HAS_TDM 1
typedef __attribute__((ext_vector_type(4))) unsigned int v4u;
typedef __attribute__((ext_vector_type(4))) int          v4i;
typedef __attribute__((ext_vector_type(8))) int          v8i;

// DMA a 128-row x 64-col bf16 tile (row stride = K elems) from global into
// LDS at byte offset lds_off, with HW padding of 16B after every 128B row
// -> LDS row stride 144B (72 bf16) for bank-conflict-free fragment reads.
__device__ __forceinline__ void tdm_load_tile(const __bf16* gsrc,
                                              unsigned lds_off,
                                              unsigned K, unsigned rows) {
    unsigned long long ga = (unsigned long long)(uintptr_t)gsrc;
    v4u g0;
    g0.x = 1u;                                       // count=1 (valid user D#)
    g0.y = lds_off;                                  // LDS byte address
    g0.z = (unsigned)(ga & 0xFFFFFFFFu);             // global_addr[31:0]
    g0.w = (unsigned)((ga >> 32) & 0x01FFFFFFu)      // global_addr[56:32]
         | (2u << 30);                               // type = 2 ("image")
    v8i g1;
    g1[0] = (int)((1u << 16)                         // data_size = 2 bytes
                | (1u << 20)                         // pad_enable
                | (4u << 22)                         // pad_interval: 32 DWORDs (128B)
                | (3u << 25));                       // pad_amount: 4 DWORDs (16B)
    g1[1] = (int)((K & 0xFFFFu) << 16);              // tensor_dim0[15:0] @ bit48
    g1[2] = (int)(((K >> 16) & 0xFFFFu)              // tensor_dim0[31:16]
                | ((rows & 0xFFFFu) << 16));         // tensor_dim1[15:0]
    g1[3] = (int)(((rows >> 16) & 0xFFFFu)           // tensor_dim1[31:16]
                | (64u << 16));                      // tile_dim0 = 64
    g1[4] = (int)128u;                               // tile_dim1 = 128 rows
    g1[5] = (int)K;                                  // tensor_dim0_stride[31:0]
    g1[6] = 0;
    g1[7] = 0;
    v4i z4 = {0, 0, 0, 0};
#if defined(__clang_major__) && (__clang_major__ >= 23)
    v8i z8 = {0, 0, 0, 0, 0, 0, 0, 0};
    __builtin_amdgcn_tensor_load_to_lds(g0, g1, z4, z4, z8, 0);
#else
    __builtin_amdgcn_tensor_load_to_lds(g0, g1, z4, z4, 0);
#endif
}
#else
#define HAS_TDM 0
#endif

// ---------------- conversion kernels (once per launch, tiny cost) ----------

__global__ void cvt_bf16_kernel(const float* __restrict__ src,
                                __bf16* __restrict__ dst, int n) {
    int i = blockIdx.x * blockDim.x + threadIdx.x;
    if (i < n) dst[i] = (__bf16)src[i];
}

// W [K][NH] f32 row-major  ->  Wt [NH][K] bf16 row-major (transposed)
__global__ void cvtT_bf16_kernel(const float* __restrict__ W,
                                 __bf16* __restrict__ Wt, int K) {
    int i = blockIdx.x * blockDim.x + threadIdx.x;   // over K*NH
    int n = i % NH_;
    int k = i / NH_;
    if (k < K) Wt[(size_t)n * K + k] = (__bf16)W[(size_t)k * NH_ + n];
}

// ---------------- WMMA GEMM:  C[M][NH] f32 = A[M][K]bf16 * Bt[NH][K]bf16^T --

template <int K>
__global__ __launch_bounds__(256)
void gemm_bf16_wmma_kernel(const __bf16* __restrict__ A,    // [M_][K]
                           const __bf16* __restrict__ Bt,   // [NH_][K]
                           float* __restrict__ C)           // [M_][NH_]
{
    constexpr int KC     = 64;      // K-chunk held in LDS
    constexpr int STR    = 72;      // LDS row stride (bf16), 144B (16B aligned)
    constexpr int NCHUNK = K / KC;
    constexpr unsigned TILE_BYTES = 128u * STR * 2u;  // 18432B per tile

    __shared__ __bf16 Asm[2][128 * STR];
    __shared__ __bf16 Bsm[2][128 * STR];

    const int tid  = threadIdx.x;
    const int lane = tid & 31;
    const int wave = tid >> 5;
    const int wm   = (wave & 3) * 32;   // wave M offset within 128-tile
    const int wn   = (wave >> 2) * 64;  // wave N offset within 128-tile
    const int m0   = blockIdx.x * 128;
    const int n0   = blockIdx.y * 128;

    const int lrow = lane & 15;         // row within 16x16 fragment
    const int hb   = (lane >> 4) * 8;   // K-half select per 16-bit WMMA layout

    const v8f vzero = {};
    v8f acc[2][4];
#pragma unroll
    for (int mi = 0; mi < 2; ++mi)
#pragma unroll
        for (int ni = 0; ni < 4; ++ni) acc[mi][ni] = vzero;

#if HAS_TDM
    // generic->LDS: low 32 bits of a __shared__ pointer are the LDS byte offset
    const unsigned ldsA = (unsigned)(uintptr_t)(&Asm[0][0]);
    const unsigned ldsB = (unsigned)(uintptr_t)(&Bsm[0][0]);

    if (wave == 0) {   // one wave drives the DMA engine; TDM ignores EXEC
        tdm_load_tile(A  + (size_t)m0 * K, ldsA, K, M_);
        tdm_load_tile(Bt + (size_t)n0 * K, ldsB, K, NH_);
    }
#endif

    for (int c = 0; c < NCHUNK; ++c) {
        const int buf = c & 1;

#if HAS_TDM
        if (wave == 0) {
            if (c + 1 < NCHUNK) {
                // prefetch next chunk into the other buffer (its consumers
                // finished before the barrier that ended iteration c-1)
                tdm_load_tile(A  + (size_t)m0 * K + (c + 1) * KC,
                              ldsA + (unsigned)(buf ^ 1) * TILE_BYTES, K, M_);
                tdm_load_tile(Bt + (size_t)n0 * K + (c + 1) * KC,
                              ldsB + (unsigned)(buf ^ 1) * TILE_BYTES, K, NH_);
                __builtin_amdgcn_s_wait_tensorcnt(2);  // current chunk landed
            } else {
                __builtin_amdgcn_s_wait_tensorcnt(0);
            }
        }
        __syncthreads();
#else
        // fallback: cooperative VGPR tile load (round-1 path)
        const int k0 = c * KC;
#pragma unroll
        for (int i = 0; i < 4; ++i) {
            int idx = tid + i * 256;
            int r   = idx >> 3;
            int v   = idx & 7;
            const uint4* ga = (const uint4*)(A  + (size_t)(m0 + r) * K + k0) + v;
            const uint4* gb = (const uint4*)(Bt + (size_t)(n0 + r) * K + k0) + v;
            *(uint4*)(&Asm[buf][r * STR + v * 8]) = *ga;
            *(uint4*)(&Bsm[buf][r * STR + v * 8]) = *gb;
        }
        __syncthreads();
#endif

        const __bf16* __restrict__ As = &Asm[buf][0];
        const __bf16* __restrict__ Bs = &Bsm[buf][0];

#pragma unroll
        for (int kk = 0; kk < KC; kk += 32) {
            v16bf afrag[2], bfrag[4];
#pragma unroll
            for (int mi = 0; mi < 2; ++mi) {
                const __bf16* p = &As[(wm + mi * 16 + lrow) * STR + kk + hb];
                v8bf lo = *(const v8bf*)p;
                v8bf hi = *(const v8bf*)(p + 16);
                afrag[mi] = __builtin_shufflevector(
                    lo, hi, 0, 1, 2, 3, 4, 5, 6, 7, 8, 9, 10, 11, 12, 13, 14, 15);
            }
#pragma unroll
            for (int ni = 0; ni < 4; ++ni) {
                const __bf16* p = &Bs[(wn + ni * 16 + lrow) * STR + kk + hb];
                v8bf lo = *(const v8bf*)p;
                v8bf hi = *(const v8bf*)(p + 16);
                bfrag[ni] = __builtin_shufflevector(
                    lo, hi, 0, 1, 2, 3, 4, 5, 6, 7, 8, 9, 10, 11, 12, 13, 14, 15);
            }
#pragma unroll
            for (int mi = 0; mi < 2; ++mi)
#pragma unroll
                for (int ni = 0; ni < 4; ++ni)
                    acc[mi][ni] = __builtin_amdgcn_wmma_f32_16x16x32_bf16(
                        false, afrag[mi], false, bfrag[ni],
                        (short)0, acc[mi][ni], false, false);
        }
        __syncthreads();   // compute(buf) done before TDM may overwrite it
    }

    // C/D layout: VGPR g -> M = g + (lane[4] ? 8 : 0); N = lane & 15
    const int rsel = (lane >> 4) * 8;
#pragma unroll
    for (int mi = 0; mi < 2; ++mi)
#pragma unroll
        for (int ni = 0; ni < 4; ++ni)
#pragma unroll
            for (int g = 0; g < 8; ++g) {
                int row = m0 + wm + mi * 16 + rsel + g;
                int col = n0 + wn + ni * 16 + (lane & 15);
                C[(size_t)row * NH_ + col] = acc[mi][ni][g];
            }
}

// ---------------- SRU scans ------------------------------------------------

__device__ __forceinline__ float sigmoidf_(float x) {
    return 1.0f / (1.0f + __expf(-x));
}
__device__ __forceinline__ float tanhf_(float x) {
    return 2.0f * sigmoidf_(2.0f * x) - 1.0f;
}

// layer-1 scan: reads U f32, writes seq1 as bf16 (direct GEMM-2 operand)
__global__ __launch_bounds__(256)
void sru_scan1_kernel(const float* __restrict__ U,
                      const float* __restrict__ bfv,
                      const float* __restrict__ brv,
                      __bf16* __restrict__ seqbf) {
    int tid = blockIdx.x * blockDim.x + threadIdx.x;  // 0 .. B*H-1
    int b = tid >> 9;            // / H_
    int h = tid & (H_ - 1);
    float bfh = bfv[h], brh = brv[h];
    float c = 0.0f;
    const float* up = U + (size_t)b * S_ * NH_;
    __bf16* op = seqbf + (size_t)b * S_ * H_;
    for (int t = 0; t < S_; ++t) {
        float xt = up[h];
        float f  = sigmoidf_(up[H_ + h] + bfh);
        float r  = sigmoidf_(up[2 * H_ + h] + brh);
        c = f * c + (1.0f - f) * xt;
        float hv = r * tanhf_(c) + (1.0f - r) * xt;
        op[h] = (__bf16)hv;
        up += NH_;
        op += H_;
    }
}

// layer-2 scan: writes d_out = [B*H final state] ++ [B*S*H hidden_all], f32
__global__ __launch_bounds__(256)
void sru_scan2_kernel(const float* __restrict__ U,
                      const float* __restrict__ bfv,
                      const float* __restrict__ brv,
                      float* __restrict__ out) {
    int tid = blockIdx.x * blockDim.x + threadIdx.x;  // 0 .. B*H-1
    int b = tid >> 9;
    int h = tid & (H_ - 1);
    float bfh = bfv[h], brh = brv[h];
    float c = 0.0f;
    const float* up = U + (size_t)b * S_ * NH_;
    float* hp = out + (size_t)B_ * H_ + (size_t)b * S_ * H_;
    float hv = 0.0f;
    for (int t = 0; t < S_; ++t) {
        float xt = up[h];
        float f  = sigmoidf_(up[H_ + h] + bfh);
        float r  = sigmoidf_(up[2 * H_ + h] + brh);
        c = f * c + (1.0f - f) * xt;
        hv = r * tanhf_(c) + (1.0f - r) * xt;
        hp[h] = hv;
        up += NH_;
        hp += H_;
    }
    out[(size_t)b * H_ + h] = hv;  // h[:, -1, :]
}

// ---------------- launcher -------------------------------------------------

extern "C" void kernel_launch(void* const* d_in, const int* in_sizes, int n_in,
                              void* d_out, int out_size, void* d_ws,
                              size_t ws_size, hipStream_t stream) {
    const float* x   = (const float*)d_in[0];   // [B,S,DIN]
    const float* W1  = (const float*)d_in[1];   // [DIN,3H]
    const float* bf1 = (const float*)d_in[2];   // [H]
    const float* br1 = (const float*)d_in[3];   // [H]
    const float* W2  = (const float*)d_in[4];   // [H,3H]
    const float* bf2 = (const float*)d_in[5];   // [H]
    const float* br2 = (const float*)d_in[6];   // [H]
    float* out = (float*)d_out;

    // workspace layout (bytes)
    char* ws = (char*)d_ws;
    float*  U      = (float*)(ws);                         // 32768*1536*4 = 192MB (reused both layers)
    __bf16* xbf    = (__bf16*)(ws + 201326592);            // 32768*256*2  = 16MB
    __bf16* seq1bf = (__bf16*)(ws + 218103808);            // 32768*512*2  = 32MB
    __bf16* Wt1    = (__bf16*)(ws + 251658240);            // 1536*256*2
    __bf16* Wt2    = (__bf16*)(ws + 252444672);            // 1536*512*2

    // 1) convert inputs / weights to bf16 (weights transposed to [N][K])
    {
        int n = M_ * DIN_;
        cvt_bf16_kernel<<<(n + 255) / 256, 256, 0, stream>>>(x, xbf, n);
    }
    {
        int n = DIN_ * NH_;
        cvtT_bf16_kernel<<<(n + 255) / 256, 256, 0, stream>>>(W1, Wt1, DIN_);
    }
    {
        int n = H_ * NH_;
        cvtT_bf16_kernel<<<(n + 255) / 256, 256, 0, stream>>>(W2, Wt2, H_);
    }

    dim3 gemm_grid(M_ / 128, NH_ / 128);   // (256, 12)

    // 2) layer 1: U = x @ W1 ; scan -> seq1 (bf16)
    gemm_bf16_wmma_kernel<DIN_><<<gemm_grid, 256, 0, stream>>>(xbf, Wt1, U);
    sru_scan1_kernel<<<(B_ * H_) / 256, 256, 0, stream>>>(U, bf1, br1, seq1bf);

    // 3) layer 2: U = seq1 @ W2 ; scan -> d_out
    gemm_bf16_wmma_kernel<H_><<<gemm_grid, 256, 0, stream>>>(seq1bf, Wt2, U);
    sru_scan2_kernel<<<(B_ * H_) / 256, 256, 0, stream>>>(U, bf2, br2, out);
}
